// cls_model_2327872274524
// MI455X (gfx1250) — matrix operand order repaired and checked by reading, not compiled, not run on hardware
//
#include <hip/hip_runtime.h>
#include <hip/hip_bf16.h>

// ---------------------------------------------------------------------------
// MI455X (gfx1250) implementation, wave32.
// All GEMM-shaped work -> one batched WMMA kernel (v_wmma_f32_16x16x32_f16).
// Operands are pre-converted to f16 in memory; B operands pre-transposed
// (N x K) so both WMMA fragments load as contiguous global_load_b128.
// ---------------------------------------------------------------------------

typedef _Float16 v16h __attribute__((ext_vector_type(16)));
typedef _Float16 v8h  __attribute__((ext_vector_type(8)));
typedef float    v8f  __attribute__((ext_vector_type(8)));

#define WAVES_PER_BLOCK 8

// Problem constants (match setup_inputs)
#define BSZ 32
#define NPTS 1024
#define RREEB 64
#define SSEG 16
#define PFPS 55
#define PPAD 64

// ---------------------------------------------------------------------------
// Batched WMMA GEMM:  C = alpha * A @ B + beta * Cin
//   A  : f16, M x K row-major (row stride lda, batch stride sA, halves)
//   Bt : f16, B TRANSPOSED, N x K row-major (row stride ldbt, halves)
//   Cin/Cout : f32
// One wave = one 16x16 C tile, K stepped by 32.
// Fragment loads are contiguous 16B segments -> global_load_b128:
//   A  lane(l): row m0+(l&15), halves [kk+8*(l>=16) ..+8) and +16 same
//   Bt lane(l): row n0+(l&15), halves [kk+8*(l>=16) ..+8) and +16 same
// If sumsq != nullptr: square-reduce the tile (wave shfl) + atomicAdd.
// Requires M%16==0, N%16==0, K%32==0; branches wave-uniform (EXEC all 1s).
// ---------------------------------------------------------------------------
__global__ void gemm_wmma_f16(const _Float16* __restrict__ A,
                              const _Float16* __restrict__ Bt,
                              const float* __restrict__ Cin,
                              float* __restrict__ Cout,
                              int M, int N, int K,
                              int lda, int ldbt, int ldcin, int ldc,
                              long sA, long sBt, long sCin, long sC,
                              float alpha, float beta, int nBatch,
                              float* sumsq)
{
    const int lane = threadIdx.x & 31;
    const int wave = threadIdx.x >> 5;
    const int tilesM = M >> 4;
    const int tilesN = N >> 4;
    const long tilesPerBatch = (long)tilesM * tilesN;
    const long total = tilesPerBatch * nBatch;
    long tile = (long)blockIdx.x * WAVES_PER_BLOCK + wave;
    if (tile >= total) return;                 // wave-uniform exit

    const int b  = (int)(tile / tilesPerBatch);
    const int t  = (int)(tile % tilesPerBatch);
    const int tm = t / tilesN;
    const int tn = t % tilesN;
    const int m0 = tm << 4;
    const int n0 = tn << 4;

    const int hl  = lane >> 4;       // half-lane group: 0 or 1
    const int l16 = lane & 15;

    const _Float16* Arow = A  + (long)b * sA  + (long)(m0 + l16) * lda  + (hl ? 8 : 0);
    const _Float16* Brow = Bt + (long)b * sBt + (long)(n0 + l16) * ldbt + (hl ? 8 : 0);

    v8f acc = {0.f,0.f,0.f,0.f,0.f,0.f,0.f,0.f};

    for (int kk = 0; kk < K; kk += 32) {
        if (kk + 32 < K) {
            __builtin_prefetch(Arow + kk + 32, 0, 1);   // global_prefetch_b8
            __builtin_prefetch(Brow + kk + 32, 0, 1);
        }
        v8h a0 = *(const v8h*)(Arow + kk);
        v8h a1 = *(const v8h*)(Arow + kk + 16);
        v8h b0 = *(const v8h*)(Brow + kk);
        v8h b1 = *(const v8h*)(Brow + kk + 16);
        v16h av, bv;
#pragma unroll
        for (int h = 0; h < 8; ++h) {
            av[h] = a0[h]; av[h + 8] = a1[h];
            bv[h] = b0[h]; bv[h + 8] = b1[h];
        }
        acc = __builtin_amdgcn_wmma_f32_16x16x32_f16(
            false, av, false, bv, (short)0, acc, false, false);
    }

    if (sumsq) {
        float s = 0.f;
#pragma unroll
        for (int r = 0; r < 8; ++r) { float v = alpha * acc[r]; s += v * v; }
        for (int off = 16; off > 0; off >>= 1)
            s += __shfl_xor(s, off, 32);
        if (lane == 0) atomicAdd(sumsq, s);
        return;
    }

    // C/D layout: VGPR r, lane l -> M = m0 + r + 8*(l/16), N = n0 + (l&15)
#pragma unroll
    for (int r = 0; r < 8; ++r) {
        const int m = m0 + r + (hl ? 8 : 0);
        const int n = n0 + l16;
        float v = alpha * acc[r];
        if (Cin) v += beta * Cin[(long)b * sCin + (long)m * ldcin + n];
        Cout[(long)b * sC + (long)m * ldc + n] = v;
    }
}

// ---------------------------------------------------------------------------
// Conversion kernels
// ---------------------------------------------------------------------------
__global__ void cvt_f16_kernel(const float* __restrict__ s, _Float16* __restrict__ d, long n)
{
    long id = (long)blockIdx.x * blockDim.x + threadIdx.x;
    if (id < n) d[id] = (_Float16)s[id];
}

// d[b, c, r] = (f16) s[b, r, c]   (R x C source -> C x R dest)
__global__ void cvtT_f16_kernel(const float* __restrict__ s, _Float16* __restrict__ d,
                                int R, int C, int lds_, long sS, int ldd, long sD, int nb)
{
    long id = (long)blockIdx.x * blockDim.x + threadIdx.x;
    long total = (long)nb * R * C;
    if (id >= total) return;
    int c = (int)(id % C);
    long t = id / C;
    int r = (int)(t % R);
    int b = (int)(t / R);
    d[(long)b * sD + (long)c * ldd + r] =
        (_Float16)s[(long)b * sS + (long)r * lds_ + c];
}

// Weight pad + transpose + f16: Wt[o, r] = (r<rows && o<cols) ? W[r, o] : 0
// Wt is padN x padK row-major.
__global__ void wpadT_f16_kernel(const float* __restrict__ W, _Float16* __restrict__ Wt,
                                 int rows, int cols, int padK, int padN)
{
    long id = (long)blockIdx.x * blockDim.x + threadIdx.x;
    if (id >= (long)padN * padK) return;
    int r = (int)(id % padK);
    int o = (int)(id / padK);
    Wt[id] = (_Float16)((r < rows && o < cols) ? W[(long)r * cols + o] : 0.f);
}

// ---------------------------------------------------------------------------
// Support kernels
// ---------------------------------------------------------------------------
__global__ void zero_kernel(float* p, long n)
{
    long id = (long)blockIdx.x * blockDim.x + threadIdx.x;
    if (id < n) p[id] = 0.f;
}

__global__ void sqnorm_kernel(const float* __restrict__ x, float* __restrict__ sq,
                              long rows, int D)
{
    long id = (long)blockIdx.x * blockDim.x + threadIdx.x;
    if (id >= rows) return;
    const float* r = x + id * D;
    float s = 0.f;
    for (int d = 0; d < D; ++d) s += r[d] * r[d];
    sq[id] = s;
}

// dist[b,i,j] = max(sq_i + sq_j - 2 x_i.x_j, 0); zero outside valid x valid
__global__ void pairdist_kernel(const float* __restrict__ x,
                                const float* __restrict__ sq,
                                float* __restrict__ dist,
                                int N, int D, int valid)
{
    int b = blockIdx.z;
    int i = blockIdx.y * 16 + threadIdx.y;
    int j = blockIdx.x * 16 + threadIdx.x;
    long base = (long)b * N;
    float v = 0.f;
    if (i < valid && j < valid) {
        const float* xi = x + (base + i) * D;
        const float* xj = x + (base + j) * D;
        float dot = 0.f;
        for (int d = 0; d < D; ++d) dot += xi[d] * xj[d];
        v = sq[base + i] + sq[base + j] - 2.f * dot;
        v = v > 0.f ? v : 0.f;
    }
    dist[(long)b * N * N + (long)i * N + j] = v;
}

// One wave per row: k-NN selection, then overwrite the row in place with the
// normalized Laplacian row L = I - A/k (row degree is exactly k).
__global__ void knn_laplacian_kernel(float* __restrict__ dist, int N,
                                     const int* __restrict__ kptr)
{
    const int lane = threadIdx.x & 31;
    const int wave = threadIdx.x >> 5;
    const long row = (long)blockIdx.x * WAVES_PER_BLOCK + wave;  // [0, B*N)
    float* drow = dist + row * N;
    const int rowIdx = (int)(row % N);
    const int k = kptr[0];

    float v[32];
#pragma unroll
    for (int j = 0; j < 32; ++j) v[j] = drow[j * 32 + lane];

    unsigned sel = 0u;
    for (int it = 0; it < k; ++it) {
        float lmin = 3.4e38f;
        int   larg = 0x7fffffff;
#pragma unroll
        for (int j = 0; j < 32; ++j) {
            int col = j * 32 + lane;
            if (v[j] < lmin) { lmin = v[j]; larg = col; }
        }
        for (int off = 16; off > 0; off >>= 1) {
            float ov = __shfl_xor(lmin, off, 32);
            int   oa = __shfl_xor(larg, off, 32);
            if (ov < lmin || (ov == lmin && oa < larg)) { lmin = ov; larg = oa; }
        }
        if ((larg & 31) == lane) {
            v[larg >> 5] = 3.4e38f;
            sel |= 1u << (larg >> 5);
        }
    }
    const float invk = 1.f / (float)k;
#pragma unroll
    for (int j = 0; j < 32; ++j) {
        int col = j * 32 + lane;
        float Lv = ((col == rowIdx) ? 1.f : 0.f)
                 - (((sel >> j) & 1u) ? invk : 0.f);
        drow[col] = Lv;
    }
}

// xpad[.,16] from x[.,7], zero padded
__global__ void pack_xpad_kernel(const float* __restrict__ x, float* __restrict__ xp,
                                 long rows)
{
    long id = (long)blockIdx.x * blockDim.x + threadIdx.x;
    if (id >= rows * 16) return;
    int c = (int)(id & 15);
    long n = id >> 4;
    xp[id] = (c < 7) ? x[n * 7 + c] : 0.f;
}

// xcat[.,32] = [x(7) | t1(7) | t2(7) | 0...]
__global__ void pack_xcat_kernel(const float* __restrict__ xp,
                                 const float* __restrict__ t1,
                                 const float* __restrict__ t2,
                                 float* __restrict__ xc, long rows)
{
    long id = (long)blockIdx.x * blockDim.x + threadIdx.x;
    if (id >= rows * 32) return;
    int c = (int)(id & 31);
    long n = id >> 5;
    float v = 0.f;
    if (c < 7)       v = xp[n * 16 + c];
    else if (c < 14) v = t1[n * 16 + (c - 7)];
    else if (c < 21) v = t2[n * 16 + (c - 14)];
    xc[id] = v;
}

// y[id] = relu(y + bias[col]) for rows < valid, else 0
__global__ void bias_relu_kernel(float* __restrict__ y, const float* __restrict__ bias,
                                 long total, int cols, int rowsMod, int valid)
{
    long id = (long)blockIdx.x * blockDim.x + threadIdx.x;
    if (id >= total) return;
    int c = (int)(id % cols);
    int r = (int)((id / cols) % rowsMod);
    float v = y[id] + bias[c];
    y[id] = (r < valid) ? (v > 0.f ? v : 0.f) : 0.f;
}

// V_reeb[b,r,f] = max_s out[b, sccs[b,r,s], f]
__global__ void segmax_reeb_kernel(const float* __restrict__ out,
                                   const int* __restrict__ sccs,
                                   float* __restrict__ V)
{
    long id = (long)blockIdx.x * blockDim.x + threadIdx.x;
    if (id >= (long)BSZ * RREEB * 64) return;
    int f = (int)(id & 63);
    int r = (int)((id >> 6) & 63);
    int b = (int)(id >> 12);
    const int* srow = sccs + ((long)b * RREEB + r) * SSEG;
    float m = -3.4e38f;
    for (int s = 0; s < SSEG; ++s) {
        int idx = srow[s];
        float v = out[((long)b * NPTS + idx) * 64 + f];
        m = v > m ? v : m;
    }
    V[id] = m;
}

// V_fps[b,r,f] (padded 55->64; pad rows zero)
__global__ void segmax_fps_kernel(const float* __restrict__ out,
                                  const int* __restrict__ sccs_fps,
                                  float* __restrict__ V)
{
    long id = (long)blockIdx.x * blockDim.x + threadIdx.x;
    if (id >= (long)BSZ * PPAD * 64) return;
    int f = (int)(id & 63);
    int r = (int)((id >> 6) & 63);
    int b = (int)(id >> 12);
    float m = 0.f;
    if (r < PFPS) {
        const int* srow = sccs_fps + ((long)b * PFPS + r) * NPTS;
        m = -3.4e38f;
        for (int n = 0; n < NPTS; ++n) {
            int idx = srow[n];
            float v = out[((long)b * NPTS + idx) * 64 + f];
            m = v > m ? v : m;
        }
    }
    V[id] = m;
}

// copy 64x64 V into cat[.,384] columns [0,64)
__global__ void copy_to_cat_kernel(const float* __restrict__ V, float* __restrict__ cat)
{
    long id = (long)blockIdx.x * blockDim.x + threadIdx.x;
    if (id >= (long)BSZ * 64 * 64) return;
    int c = (int)(id & 63);
    int r = (int)((id >> 6) & 63);
    int b = (int)(id >> 12);
    cat[((long)b * 64 + r) * 384 + c] = V[id];
}

// dinv[b,i] = d>0 ? rsqrt(max(d,1e-12)) : 0 with d = sum_{j<valid} dist[b,i,j]
__global__ void dinv_kernel(const float* __restrict__ dist, float* __restrict__ dinv,
                            int N, int valid)
{
    long id = (long)blockIdx.x * blockDim.x + threadIdx.x;
    if (id >= (long)BSZ * N) return;
    int i = (int)(id % N);
    int b = (int)(id / N);
    float dv = 0.f;
    if (i < valid) {
        const float* row = dist + ((long)b * N + i) * N;
        float d = 0.f;
        for (int j = 0; j < valid; ++j) d += row[j];
        float dm = d > 1e-12f ? d : 1e-12f;
        dv = (d > 0.f) ? rsqrtf(dm) : 0.f;
    }
    dinv[id] = dv;
}

// In-place: L = I - dist * dinv_i * dinv_j inside valid x valid, 0 outside
__global__ void build_lap_kernel(float* __restrict__ dist, const float* __restrict__ dinv,
                                 int N, int valid)
{
    long id = (long)blockIdx.x * blockDim.x + threadIdx.x;
    if (id >= (long)BSZ * N * N) return;
    int j = (int)(id % N);
    int i = (int)((id / N) % N);
    int b = (int)(id / ((long)N * N));
    float v = 0.f;
    if (i < valid && j < valid) {
        v = ((i == j) ? 1.f : 0.f)
          - dist[id] * dinv[(long)b * N + i] * dinv[(long)b * N + j];
    }
    dist[id] = v;
}

// feat[b, 0:256] = max_n out_reeb ; feat[b, 256:512] = max_n out_fps
__global__ void maxpool_feat_kernel(const float* __restrict__ oreeb,
                                    const float* __restrict__ ofps,
                                    float* __restrict__ feat)
{
    long id = (long)blockIdx.x * blockDim.x + threadIdx.x;
    if (id >= (long)BSZ * 512) return;
    int c = (int)(id & 511);
    int b = (int)(id >> 9);
    const float* src = (c < 256) ? (oreeb + (long)b * 64 * 256 + c)
                                 : (ofps  + (long)b * 64 * 256 + (c - 256));
    float m = -3.4e38f;
    for (int n = 0; n < 64; ++n) {
        float v = src[(long)n * 256];
        m = v > m ? v : m;
    }
    feat[id] = m;
}

__global__ void logits_kernel(const float* __restrict__ lp, const float* __restrict__ b3,
                              float* __restrict__ outp)
{
    long id = (long)blockIdx.x * blockDim.x + threadIdx.x;
    if (id >= (long)BSZ * 40) return;
    int o = (int)(id % 40);
    int b = (int)(id / 40);
    outp[id] = lp[(long)b * 48 + o] + b3[o];
}

__global__ void small_regs_kernel(const float* __restrict__ w1,
                                  const float* __restrict__ b1,
                                  const float* __restrict__ w3,
                                  const float* __restrict__ b3,
                                  float* __restrict__ outp)
{
    if (threadIdx.x != 0 || blockIdx.x != 0) return;
    float s = 0.f;
    for (int i = 0; i < 512; ++i) { float v = w1[(long)i * 256]; s += v * v; }
    outp[BSZ * 40 + 3] = s;
    outp[BSZ * 40 + 4] = b1[0] * b1[0];
    s = 0.f;
    for (int i = 0; i < 256; ++i) { float v = w3[(long)i * 40]; s += v * v; }
    outp[BSZ * 40 + 5] = s;
    outp[BSZ * 40 + 6] = b3[0] * b3[0];
}

// ---------------------------------------------------------------------------
// Host-side helpers
// ---------------------------------------------------------------------------
static inline int g1(long n, int bs) { return (int)((n + bs - 1) / bs); }

static inline void gemm(hipStream_t s,
                        const _Float16* A, const _Float16* Bt,
                        const float* Cin, float* C,
                        int M, int N, int K,
                        int lda, int ldbt, int ldcin, int ldc,
                        long sA, long sBt, long sCin, long sC,
                        float alpha, float beta, int nb, float* sumsq = nullptr)
{
    long tiles = (long)(M / 16) * (N / 16) * nb;
    int blocks = (int)((tiles + WAVES_PER_BLOCK - 1) / WAVES_PER_BLOCK);
    gemm_wmma_f16<<<blocks, 32 * WAVES_PER_BLOCK, 0, s>>>(
        A, Bt, Cin, C, M, N, K, lda, ldbt, ldcin, ldc, sA, sBt, sCin, sC,
        alpha, beta, nb, sumsq);
}

static inline void cvt(hipStream_t s, const float* src, _Float16* dst, long n)
{
    cvt_f16_kernel<<<g1(n, 256), 256, 0, s>>>(src, dst, n);
}

static inline void cvtT(hipStream_t s, const float* src, _Float16* dst,
                        int R, int C, int lds_, long sS, int ldd, long sD, int nb)
{
    cvtT_f16_kernel<<<g1((long)nb * R * C, 256), 256, 0, s>>>(
        src, dst, R, C, lds_, sS, ldd, sD, nb);
}

// ---------------------------------------------------------------------------
extern "C" void kernel_launch(void* const* d_in, const int* in_sizes, int n_in,
                              void* d_out, int out_size, void* d_ws, size_t ws_size,
                              hipStream_t stream)
{
    const float* x       = (const float*)d_in[0];
    // d_in[1] (x2) unused by the reference
    const float* Lreeb   = (const float*)d_in[2];
    const float* W1      = (const float*)d_in[3];
    const float* b1      = (const float*)d_in[4];
    const float* W_reeb  = (const float*)d_in[5];
    const float* b_reeb  = (const float*)d_in[6];
    const float* W_fps   = (const float*)d_in[7];
    const float* b_fps   = (const float*)d_in[8];
    const float* fc1_w   = (const float*)d_in[9];
    const float* fc1_b   = (const float*)d_in[10];
    const float* fc3_w   = (const float*)d_in[11];
    const float* fc3_b   = (const float*)d_in[12];
    const int*   sccs    = (const int*)d_in[13];
    const int*   sccsf   = (const int*)d_in[14];
    const int*   kptr    = (const int*)d_in[15];

    float* out = (float*)d_out;

    // ---- workspace layout ----
    float* ws = (float*)d_ws;
    size_t off = 0;
    auto allocF = [&](size_t n) {
        off = (off + 15) & ~(size_t)15; float* p = ws + off; off += n; return p;
    };
    auto allocH = [&](size_t n) {
        off = (off + 15) & ~(size_t)15;
        _Float16* p = (_Float16*)(ws + off); off += (n + 1) / 2; return p;
    };
    // f32 buffers
    float* Lbuf  = allocF((size_t)BSZ * NPTS * NPTS);   // dist -> Laplacian
    float* xpad  = allocF((size_t)BSZ * NPTS * 16);
    float* t1    = allocF((size_t)BSZ * NPTS * 16);
    float* t2    = allocF((size_t)BSZ * NPTS * 16);
    float* xcat  = allocF((size_t)BSZ * NPTS * 32);
    float* out1  = allocF((size_t)BSZ * NPTS * 64);
    float* T1    = allocF((size_t)BSZ * NPTS * 64);
    float* sq    = allocF((size_t)BSZ * NPTS);
    float* Vreeb = allocF((size_t)BSZ * 64 * 64);
    float* rcat  = allocF((size_t)BSZ * 64 * 384);
    float* oreeb = allocF((size_t)BSZ * 64 * 256);
    float* Treeb = allocF((size_t)BSZ * 64 * 256);
    float* Vfps  = allocF((size_t)BSZ * 64 * 64);
    float* fsq   = allocF((size_t)BSZ * 64);
    float* fdinv = allocF((size_t)BSZ * 64);
    float* Lfps  = allocF((size_t)BSZ * 64 * 64);
    float* fcat  = allocF((size_t)BSZ * 64 * 384);
    float* ofps  = allocF((size_t)BSZ * 64 * 256);
    float* Tfps  = allocF((size_t)BSZ * 64 * 256);
    float* feat  = allocF((size_t)BSZ * 512);
    float* hbuf  = allocF((size_t)BSZ * 256);
    float* lpad  = allocF((size_t)BSZ * 48);
    // f16 buffers (A row-major M x K, or B transposed N x K)
    _Float16* Lh     = allocH((size_t)BSZ * NPTS * NPTS);
    _Float16* xpadT  = allocH((size_t)BSZ * 16 * NPTS);
    _Float16* t1T    = allocH((size_t)BSZ * 16 * NPTS);
    _Float16* xcatH  = allocH((size_t)BSZ * NPTS * 32);
    _Float16* W1T    = allocH(64 * 32);
    _Float16* out1T  = allocH((size_t)BSZ * 64 * NPTS);
    _Float16* T1T    = allocH((size_t)BSZ * 64 * NPTS);
    _Float16* LreebH = allocH((size_t)BSZ * 64 * 64);
    _Float16* rcatT  = allocH((size_t)BSZ * 384 * 64);
    _Float16* rcatH  = allocH((size_t)BSZ * 64 * 384);
    _Float16* WreebT = allocH(256 * 384);
    _Float16* oreebT = allocH((size_t)BSZ * 256 * 64);
    _Float16* TreebT = allocH((size_t)BSZ * 256 * 64);
    _Float16* LfpsH  = allocH((size_t)BSZ * 64 * 64);
    _Float16* fcatT  = allocH((size_t)BSZ * 384 * 64);
    _Float16* fcatH  = allocH((size_t)BSZ * 64 * 384);
    _Float16* WfpsT  = allocH(256 * 384);
    _Float16* ofpsT  = allocH((size_t)BSZ * 256 * 64);
    _Float16* TfpsT  = allocH((size_t)BSZ * 256 * 64);
    _Float16* featH  = allocH((size_t)BSZ * 512);
    _Float16* fc1wT  = allocH(256 * 512);
    _Float16* hH     = allocH((size_t)BSZ * 256);
    _Float16* W3T    = allocH(48 * 256);
    (void)ws_size;

    const int BT = 256;
    float* regs = out + BSZ * 40;   // [reg1, reg2, reg3, ...]
    zero_kernel<<<1, 32, 0, stream>>>(regs, 3);   // atomically accumulated

    // ---- KNN Laplacian: dist -> topk -> L = I - A/k (in place), then f16 ----
    sqnorm_kernel<<<g1((long)BSZ * NPTS, BT), BT, 0, stream>>>(x, sq, (long)BSZ * NPTS, 7);
    {
        dim3 grid(NPTS / 16, NPTS / 16, BSZ), blk(16, 16);
        pairdist_kernel<<<grid, blk, 0, stream>>>(x, sq, Lbuf, NPTS, 7, NPTS);
    }
    knn_laplacian_kernel<<<(BSZ * NPTS) / WAVES_PER_BLOCK, 32 * WAVES_PER_BLOCK, 0, stream>>>(
        Lbuf, NPTS, kptr);
    cvt(stream, Lbuf, Lh, (long)BSZ * NPTS * NPTS);

    // ---- Cheb conv #1 (K=3): feature dim 7->16, concat 21->32 ----
    pack_xpad_kernel<<<g1((long)BSZ * NPTS * 16, BT), BT, 0, stream>>>(x, xpad, (long)BSZ * NPTS);
    cvtT(stream, xpad, xpadT, NPTS, 16, 16, (long)NPTS * 16, NPTS, (long)16 * NPTS, BSZ);
    // t1 = L @ xpad
    gemm(stream, Lh, xpadT, nullptr, t1, NPTS, 16, NPTS,
         NPTS, NPTS, 0, 16, (long)NPTS * NPTS, (long)16 * NPTS, 0, (long)NPTS * 16,
         1.f, 0.f, BSZ);
    cvtT(stream, t1, t1T, NPTS, 16, 16, (long)NPTS * 16, NPTS, (long)16 * NPTS, BSZ);
    // t2 = 2*L@t1 - xpad
    gemm(stream, Lh, t1T, xpad, t2, NPTS, 16, NPTS,
         NPTS, NPTS, 16, 16, (long)NPTS * NPTS, (long)16 * NPTS, (long)NPTS * 16, (long)NPTS * 16,
         2.f, -1.f, BSZ);
    pack_xcat_kernel<<<g1((long)BSZ * NPTS * 32, BT), BT, 0, stream>>>(
        xpad, t1, t2, xcat, (long)BSZ * NPTS);
    cvt(stream, xcat, xcatH, (long)BSZ * NPTS * 32);
    wpadT_f16_kernel<<<g1(64 * 32, BT), BT, 0, stream>>>(W1, W1T, 21, 64, 32, 64);
    // out1 = relu(xcat @ W1 + b1)
    gemm(stream, xcatH, W1T, nullptr, out1, NPTS, 64, 32,
         32, 32, 0, 64, (long)NPTS * 32, 0, 0, (long)NPTS * 64, 1.f, 0.f, BSZ);
    bias_relu_kernel<<<g1((long)BSZ * NPTS * 64, BT), BT, 0, stream>>>(
        out1, b1, (long)BSZ * NPTS * 64, 64, NPTS, NPTS);
    cvtT(stream, out1, out1T, NPTS, 64, 64, (long)NPTS * 64, NPTS, (long)64 * NPTS, BSZ);

    // ---- reg1 = || out1^T (L @ out1) ||_F^2 ----
    gemm(stream, Lh, out1T, nullptr, T1, NPTS, 64, NPTS,
         NPTS, NPTS, 0, 64, (long)NPTS * NPTS, (long)64 * NPTS, 0, (long)NPTS * 64,
         1.f, 0.f, BSZ);
    cvtT(stream, T1, T1T, NPTS, 64, 64, (long)NPTS * 64, NPTS, (long)64 * NPTS, BSZ);
    gemm(stream, out1T, T1T, nullptr, nullptr, 64, 64, NPTS,
         NPTS, NPTS, 0, 0, (long)64 * NPTS, (long)64 * NPTS, 0, 0, 1.f, 0.f, BSZ, regs + 0);

    // ---- Reeb branch ----
    cvt(stream, Lreeb, LreebH, (long)BSZ * 64 * 64);
    segmax_reeb_kernel<<<g1((long)BSZ * 64 * 64, BT), BT, 0, stream>>>(out1, sccs, Vreeb);
    copy_to_cat_kernel<<<g1((long)BSZ * 64 * 64, BT), BT, 0, stream>>>(Vreeb, rcat);
    cvtT(stream, rcat, rcatT, 64, 64, 384, (long)64 * 384, 64, (long)384 * 64, BSZ);
    for (int kc = 1; kc < 6; ++kc) {
        const float* cin = (kc >= 2) ? (rcat + (kc - 2) * 64) : nullptr;
        float alpha = (kc >= 2) ? 2.f : 1.f, beta = (kc >= 2) ? -1.f : 0.f;
        gemm(stream, LreebH, rcatT + (long)(kc - 1) * 64 * 64, cin, rcat + kc * 64,
             64, 64, 64, 64, 64, 384, 384,
             (long)64 * 64, (long)384 * 64, (long)64 * 384, (long)64 * 384, alpha, beta, BSZ);
        cvtT(stream, rcat + kc * 64, rcatT + (long)kc * 64 * 64,
             64, 64, 384, (long)64 * 384, 64, (long)384 * 64, BSZ);
    }
    cvt(stream, rcat, rcatH, (long)BSZ * 64 * 384);
    wpadT_f16_kernel<<<g1(256 * 384, BT), BT, 0, stream>>>(W_reeb, WreebT, 384, 256, 384, 256);
    gemm(stream, rcatH, WreebT, nullptr, oreeb, 64, 256, 384,
         384, 384, 0, 256, (long)64 * 384, 0, 0, (long)64 * 256, 1.f, 0.f, BSZ);
    bias_relu_kernel<<<g1((long)BSZ * 64 * 256, BT), BT, 0, stream>>>(
        oreeb, b_reeb, (long)BSZ * 64 * 256, 256, 64, 64);
    cvtT(stream, oreeb, oreebT, 64, 256, 256, (long)64 * 256, 64, (long)256 * 64, BSZ);
    // reg2
    gemm(stream, LreebH, oreebT, nullptr, Treeb, 64, 256, 64,
         64, 64, 0, 256, (long)64 * 64, (long)256 * 64, 0, (long)64 * 256, 1.f, 0.f, BSZ);
    cvtT(stream, Treeb, TreebT, 64, 256, 256, (long)64 * 256, 64, (long)256 * 64, BSZ);
    gemm(stream, oreebT, TreebT, nullptr, nullptr, 256, 256, 64,
         64, 64, 0, 0, (long)256 * 64, (long)256 * 64, 0, 0, 1.f, 0.f, BSZ, regs + 1);

    // ---- FPS branch (P=55 padded to 64; pad rows stay exactly 0) ----
    segmax_fps_kernel<<<g1((long)BSZ * PPAD * 64, BT), BT, 0, stream>>>(out1, sccsf, Vfps);
    sqnorm_kernel<<<g1((long)BSZ * PPAD, BT), BT, 0, stream>>>(Vfps, fsq, (long)BSZ * PPAD, 64);
    {
        dim3 grid(PPAD / 16, PPAD / 16, BSZ), blk(16, 16);
        pairdist_kernel<<<grid, blk, 0, stream>>>(Vfps, fsq, Lfps, PPAD, 64, PFPS);
    }
    dinv_kernel<<<g1((long)BSZ * PPAD, BT), BT, 0, stream>>>(Lfps, fdinv, PPAD, PFPS);
    build_lap_kernel<<<g1((long)BSZ * PPAD * PPAD, BT), BT, 0, stream>>>(Lfps, fdinv, PPAD, PFPS);
    cvt(stream, Lfps, LfpsH, (long)BSZ * 64 * 64);
    copy_to_cat_kernel<<<g1((long)BSZ * 64 * 64, BT), BT, 0, stream>>>(Vfps, fcat);
    cvtT(stream, fcat, fcatT, 64, 64, 384, (long)64 * 384, 64, (long)384 * 64, BSZ);
    for (int kc = 1; kc < 6; ++kc) {
        const float* cin = (kc >= 2) ? (fcat + (kc - 2) * 64) : nullptr;
        float alpha = (kc >= 2) ? 2.f : 1.f, beta = (kc >= 2) ? -1.f : 0.f;
        gemm(stream, LfpsH, fcatT + (long)(kc - 1) * 64 * 64, cin, fcat + kc * 64,
             64, 64, 64, 64, 64, 384, 384,
             (long)64 * 64, (long)384 * 64, (long)64 * 384, (long)64 * 384, alpha, beta, BSZ);
        cvtT(stream, fcat + kc * 64, fcatT + (long)kc * 64 * 64,
             64, 64, 384, (long)64 * 384, 64, (long)384 * 64, BSZ);
    }
    cvt(stream, fcat, fcatH, (long)BSZ * 64 * 384);
    wpadT_f16_kernel<<<g1(256 * 384, BT), BT, 0, stream>>>(W_fps, WfpsT, 384, 256, 384, 256);
    gemm(stream, fcatH, WfpsT, nullptr, ofps, 64, 256, 384,
         384, 384, 0, 256, (long)64 * 384, 0, 0, (long)64 * 256, 1.f, 0.f, BSZ);
    bias_relu_kernel<<<g1((long)BSZ * 64 * 256, BT), BT, 0, stream>>>(
        ofps, b_fps, (long)BSZ * 64 * 256, 256, 64, PFPS);
    cvtT(stream, ofps, ofpsT, 64, 256, 256, (long)64 * 256, 64, (long)256 * 64, BSZ);
    // reg3
    gemm(stream, LfpsH, ofpsT, nullptr, Tfps, 64, 256, 64,
         64, 64, 0, 256, (long)64 * 64, (long)256 * 64, 0, (long)64 * 256, 1.f, 0.f, BSZ);
    cvtT(stream, Tfps, TfpsT, 64, 256, 256, (long)64 * 256, 64, (long)256 * 64, BSZ);
    gemm(stream, ofpsT, TfpsT, nullptr, nullptr, 256, 256, 64,
         64, 64, 0, 0, (long)256 * 64, (long)256 * 64, 0, 0, 1.f, 0.f, BSZ, regs + 2);

    // ---- head: maxpool -> fc1 -> relu -> fc3 ----
    maxpool_feat_kernel<<<g1((long)BSZ * 512, BT), BT, 0, stream>>>(oreeb, ofps, feat);
    cvt(stream, feat, featH, (long)BSZ * 512);
    wpadT_f16_kernel<<<g1(256 * 512, BT), BT, 0, stream>>>(fc1_w, fc1wT, 512, 256, 512, 256);
    gemm(stream, featH, fc1wT, nullptr, hbuf, BSZ, 256, 512,
         512, 512, 0, 256, 0, 0, 0, 0, 1.f, 0.f, 1);
    bias_relu_kernel<<<g1((long)BSZ * 256, BT), BT, 0, stream>>>(
        hbuf, fc1_b, (long)BSZ * 256, 256, BSZ, BSZ);
    cvt(stream, hbuf, hH, (long)BSZ * 256);
    wpadT_f16_kernel<<<g1(48 * 256, BT), BT, 0, stream>>>(fc3_w, W3T, 256, 40, 256, 48);
    gemm(stream, hH, W3T, nullptr, lpad, BSZ, 48, 256,
         256, 256, 0, 48, 0, 0, 0, 0, 1.f, 0.f, 1);
    logits_kernel<<<g1((long)BSZ * 40, BT), BT, 0, stream>>>(lpad, fc3_b, out);

    small_regs_kernel<<<1, 1, 0, stream>>>(fc1_w, fc1_b, fc3_w, fc3_b, out);
    (void)in_sizes; (void)n_in; (void)out_size;
}
